// HeteroGNN_9053791060419
// MI455X (gfx1250) — compile-verified, matrix operand order checked by reading, not run on hardware
//
#include <hip/hip_runtime.h>
#include <math.h>

// ---------------------------------------------------------------------------
// HeteroGNN (2x HeteroConv of single-head GATConv, H=64) for MI455X / gfx1250.
//
// Exact algebraic reductions vs. the literal reference:
//  * (eh * ae).sum(-1)  ==  eah @ (We @ ae)        -> per-edge GEMV "ael"
//  * bi-GAT a_dst       ==  x_dst @ (Wd @ ad)      -> per-node GEMV
//  * self-loop logit offset == mean_e(ael[e])  (scalar), message == w_self*h[i]
// Dense work: [n x 64] @ [64 x 64] node transforms -> v_wmma_f32_16x16x32_bf16
// (bf16 MAC, f32 accumulate); scatter/softmax stages are f32 + L2 atomics.
// ---------------------------------------------------------------------------

typedef __attribute__((ext_vector_type(16))) __bf16 v16bf;
typedef __attribute__((ext_vector_type(8)))  float  v8f;

__device__ __forceinline__ float lrelu(float x, float s) { return x > 0.0f ? x : s * x; }

__device__ __forceinline__ void atomicMaxFloat(float* addr, float val) {
  // monotonic float-max via integer atomics (works for mixed signs)
  if (val >= 0.0f) atomicMax((int*)addr, __float_as_int(val));
  else             atomicMin((unsigned int*)addr, (unsigned int)__float_as_int(val));
}

// ---------------------------------------------------------------------------
// WMMA GEMM: Y[N x 64] = act(X[N x K] @ W[K x 64] + bias), bf16 MAC / f32 acc.
// KCHUNKS = ceil(K/32). Block = 256 threads = 8 waves; each wave owns a
// 16-row x 64-col output tile (4 x v_wmma_f32_16x16x32_bf16 per K-chunk).
// ---------------------------------------------------------------------------
template<int KCHUNKS, int ACT, bool BIAS>
__global__ __launch_bounds__(256)
void k_gemm_n64(const float* __restrict__ X, const float* __restrict__ W,
                const float* __restrict__ bias, float* __restrict__ Y,
                int N, int K) {
  constexpr int KPAD = KCHUNKS * 32;
  __shared__ __bf16 Wt[64 * KPAD];              // W transposed: Wt[col][k], zero padded
  for (int idx = threadIdx.x; idx < 64 * KPAD; idx += 256) {
    int col = idx / KPAD, k = idx % KPAD;
    Wt[idx] = (__bf16)((k < K) ? W[k * 64 + col] : 0.0f);
  }
  __syncthreads();

  const int lane  = threadIdx.x & 31;
  const int wave  = threadIdx.x >> 5;
  const int rowb  = blockIdx.x * 128 + wave * 16;
  if (rowb >= N) return;                         // wave-uniform exit (EXEC stays full)
  int row = rowb + (lane & 15);
  if (row >= N) row = N - 1;                     // clamp loads; stores are guarded
  const float* Xrow = X + (long)row * K;
  const int khalf = lane >> 4;                   // 16-bit A/B layout half select

  v8f acc[4] = {v8f{}, v8f{}, v8f{}, v8f{}};
#pragma unroll
  for (int kc = 0; kc < KCHUNKS; ++kc) {
    const int kbase = kc * 32 + khalf * 8;
    v16bf a;
    // A: lane<16 holds K{kb..kb+7, kb+16..kb+23} with kb=kc*32; lane>=16: +8
#pragma unroll
    for (int half = 0; half < 2; ++half) {
      const int kb = kbase + 16 * half;
      if (kb + 7 < K) {                          // vectorized in-range path (K=32/64)
        const float4 f0 = *(const float4*)(Xrow + kb);
        const float4 f1 = *(const float4*)(Xrow + kb + 4);
        a[8 * half + 0] = (__bf16)f0.x; a[8 * half + 1] = (__bf16)f0.y;
        a[8 * half + 2] = (__bf16)f0.z; a[8 * half + 3] = (__bf16)f0.w;
        a[8 * half + 4] = (__bf16)f1.x; a[8 * half + 5] = (__bf16)f1.y;
        a[8 * half + 6] = (__bf16)f1.z; a[8 * half + 7] = (__bf16)f1.w;
      } else {                                   // zero-padded tail (K=12/16)
#pragma unroll
        for (int e = 0; e < 8; ++e)
          a[8 * half + e] = (__bf16)((kb + e < K) ? Xrow[kb + e] : 0.0f);
      }
    }
#pragma unroll
    for (int t = 0; t < 4; ++t) {
      const int col = t * 16 + (lane & 15);
      const __bf16* p = Wt + col * KPAD + kc * 32 + khalf * 8;
      v16bf b;
#pragma unroll
      for (int e = 0; e < 8; ++e) { b[e] = p[e]; b[e + 8] = p[16 + e]; }
      acc[t] = __builtin_amdgcn_wmma_f32_16x16x32_bf16(
          false, a, false, b, (short)0, acc[t], false, false);
    }
  }
  // D layout: VGPR r -> row rowb + r + khalf*8, col = tile*16 + (lane&15)
  const int rrow0 = rowb + khalf * 8;
  if (rowb + 16 <= N) {
    // fast path (wave-uniform): full tile in range, unguarded coalesced stores
#pragma unroll
    for (int t = 0; t < 4; ++t) {
      const int col = t * 16 + (lane & 15);
      const float bv = BIAS ? bias[col] : 0.0f;
#pragma unroll
      for (int r = 0; r < 8; ++r) {
        float v = acc[t][r] + bv;
        if (ACT == 1) v = lrelu(v, 0.01f);
        Y[(long)(rrow0 + r) * 64 + col] = v;
      }
    }
  } else {
#pragma unroll
    for (int t = 0; t < 4; ++t) {
      const int col = t * 16 + (lane & 15);
      const float bv = BIAS ? bias[col] : 0.0f;
#pragma unroll
      for (int r = 0; r < 8; ++r) {
        int orow = rrow0 + r;
        if (orow < N) {
          float v = acc[t][r] + bv;
          if (ACT == 1) v = lrelu(v, 0.01f);
          Y[(long)orow * 64 + col] = v;
        }
      }
    }
  }
}

// y[64] = M[64x64] @ v[64]   (one 64-thread block)
__global__ void k_matvec64(const float* __restrict__ M, const float* __restrict__ v,
                           float* __restrict__ y) {
  __shared__ float vs[64];
  if (threadIdx.x < 64) vs[threadIdx.x] = v[threadIdx.x];
  __syncthreads();
  int i = threadIdx.x;
  if (i < 64) {
    float acc = 0.0f;
#pragma unroll
    for (int j = 0; j < 64; ++j) acc += M[i * 64 + j] * vs[j];
    y[i] = acc;
  }
}

// per-row dots: o1[i]=X[i]·v1, o2[i]=X[i]·v2  (wave per row)
__global__ __launch_bounds__(256)
void k_rowdot2(const float* __restrict__ X, const float* __restrict__ v1,
               const float* __restrict__ v2, float* __restrict__ o1,
               float* __restrict__ o2, int N) {
  int lane = threadIdx.x & 31;
  int row  = blockIdx.x * 8 + (threadIdx.x >> 5);
  if (row >= N) return;
  const float* xr = X + (long)row * 64;
  float x0 = xr[lane], x1 = xr[32 + lane];
  float d1 = x0 * v1[lane] + x1 * v1[32 + lane];
  float d2 = x0 * v2[lane] + x1 * v2[32 + lane];
#pragma unroll
  for (int off = 16; off > 0; off >>= 1) {
    d1 += __shfl_xor(d1, off, 32);
    d2 += __shfl_xor(d2, off, 32);
  }
  if (lane == 0) { o1[row] = d1; o2[row] = d2; }
}

__global__ __launch_bounds__(256)
void k_rowdot1(const float* __restrict__ X, const float* __restrict__ v,
               float* __restrict__ o, int N) {
  int lane = threadIdx.x & 31;
  int row  = blockIdx.x * 8 + (threadIdx.x >> 5);
  if (row >= N) return;
  const float* xr = X + (long)row * 64;
  float d = xr[lane] * v[lane] + xr[32 + lane] * v[32 + lane];
#pragma unroll
  for (int off = 16; off > 0; off >>= 1) d += __shfl_xor(d, off, 32);
  if (lane == 0) o[row] = d;
}

// ael[e] = leakyrelu_0.01(ea[e] @ Wp + bp) · wae ; optional sum for c_self
__global__ __launch_bounds__(256)
void k_edge_ael(const float* __restrict__ EA, const float* __restrict__ Wp,
                const float* __restrict__ bp, const float* __restrict__ wae,
                float* __restrict__ ael, float* __restrict__ csum, int E, int De) {
  __shared__ float sW[12 * 64];
  __shared__ float sB[64];
  __shared__ float sV[64];
  __shared__ float red[256];
  for (int idx = threadIdx.x; idx < De * 64; idx += 256) sW[idx] = Wp[idx];
  if (threadIdx.x < 64) { sB[threadIdx.x] = bp[threadIdx.x]; sV[threadIdx.x] = wae[threadIdx.x]; }
  __syncthreads();
  int e = blockIdx.x * 256 + threadIdx.x;
  float acc = 0.0f;
  if (e < E) {
    float ea[12];
    for (int k = 0; k < De; ++k) ea[k] = EA[(long)e * De + k];
#pragma unroll 4
    for (int j = 0; j < 64; ++j) {
      float t = sB[j];
      for (int k = 0; k < De; ++k) t += ea[k] * sW[k * 64 + j];
      acc += lrelu(t, 0.01f) * sV[j];
    }
    ael[e] = acc;
  }
  if (csum) {
    red[threadIdx.x] = acc;
    __syncthreads();
    for (int s = 128; s > 0; s >>= 1) {
      if (threadIdx.x < (unsigned)s) red[threadIdx.x] += red[threadIdx.x + s];
      __syncthreads();
    }
    if (threadIdx.x == 0) atomicAdd(csum, red[0]);
  }
}

// same-GAT: m[i] = self-loop logit = lrelu(asrc+adst+c_self, 0.2)
__global__ void k_node_init_same(const float* __restrict__ asrc, const float* __restrict__ adst,
                                 const float* __restrict__ csum, float invE,
                                 float* __restrict__ m, int n) {
  int i = blockIdx.x * 256 + threadIdx.x;
  if (i >= n) return;
  m[i] = lrelu(asrc[i] + adst[i] + csum[0] * invE, 0.2f);
}

__global__ void k_node_init_bi(float* __restrict__ m, float* __restrict__ s, int n) {
  int i = blockIdx.x * 256 + threadIdx.x;
  if (i >= n) return;
  m[i] = -__builtin_inff();
  s[i] = 0.0f;
}

__global__ void k_edge_logit_max(const int* __restrict__ src, const int* __restrict__ dst,
                                 const float* __restrict__ ael, const float* __restrict__ asrc,
                                 const float* __restrict__ adst, float* __restrict__ logit,
                                 float* __restrict__ m, int E) {
  int e = blockIdx.x * 256 + threadIdx.x;
  if (e >= E) return;
  float l = lrelu(asrc[src[e]] + adst[dst[e]] + ael[e], 0.2f);
  logit[e] = l;
  atomicMaxFloat(m + dst[e], l);
}

// same-GAT: s[i] = exp(lself - m[i])  (overwrite; edges atomicAdd after)
__global__ void k_node_s_self(const float* __restrict__ asrc, const float* __restrict__ adst,
                              const float* __restrict__ csum, float invE,
                              const float* __restrict__ m, float* __restrict__ s, int n) {
  int i = blockIdx.x * 256 + threadIdx.x;
  if (i >= n) return;
  float l = lrelu(asrc[i] + adst[i] + csum[0] * invE, 0.2f);
  s[i] = __expf(l - m[i]);
}

__global__ void k_edge_expsum(const int* __restrict__ dst, const float* __restrict__ logit,
                              const float* __restrict__ m, float* __restrict__ s, int E) {
  int e = blockIdx.x * 256 + threadIdx.x;
  if (e >= E) return;
  int d = dst[e];
  atomicAdd(s + d, __expf(logit[e] - m[d]));
}

// finalize per-edge softmax coefficient IN PLACE: logit[e] <- exp(l-m)/s
__global__ void k_edge_coef(const int* __restrict__ dst, float* __restrict__ logit,
                            const float* __restrict__ m, const float* __restrict__ s, int E) {
  int e = blockIdx.x * 256 + threadIdx.x;
  if (e >= E) return;
  int d = dst[e];
  logit[e] = __expf(logit[e] - m[d]) / s[d];
}

// scatter: xnext[dst,:] += coef[e] * h[src,:]   (64 threads per edge)
__global__ __launch_bounds__(256)
void k_edge_agg(const int* __restrict__ src, const int* __restrict__ dst,
                const float* __restrict__ coef, const float* __restrict__ h,
                float* __restrict__ xnext, int E) {
  long t = (long)blockIdx.x * 256 + threadIdx.x;
  if (t >= (long)E * 64) return;
  int e = (int)(t >> 6), c = (int)(t & 63);
  atomicAdd(xnext + (long)dst[e] * 64 + c, coef[e] * h[(long)src[e] * 64 + c]);
}

// same-GAT epilogue: self-loop message + conv bias
__global__ void k_node_finish_same(const float* __restrict__ h, const float* __restrict__ asrc,
                                   const float* __restrict__ adst, const float* __restrict__ csum,
                                   float invE, const float* __restrict__ m,
                                   const float* __restrict__ s, const float* __restrict__ bias,
                                   float* __restrict__ xnext, int n) {
  long t = (long)blockIdx.x * 256 + threadIdx.x;
  if (t >= (long)n * 64) return;
  int i = (int)(t >> 6), c = (int)(t & 63);
  float l = lrelu(asrc[i] + adst[i] + csum[0] * invE, 0.2f);
  float coef = __expf(l - m[i]) / s[i];
  xnext[t] += coef * h[t] + bias[c];
}

__global__ void k_node_bias(const float* __restrict__ bias, float* __restrict__ xnext, int n) {
  long t = (long)blockIdx.x * 256 + threadIdx.x;
  if (t >= (long)n * 64) return;
  xnext[t] += bias[t & 63];
}

__global__ void k_zero(float* __restrict__ p, long n) {
  long t = (long)blockIdx.x * 256 + threadIdx.x;
  if (t < n) p[t] = 0.0f;
}

__global__ void k_out(const float* __restrict__ X, const float* __restrict__ w,
                      const float* __restrict__ b, float* __restrict__ out, int N) {
  int lane = threadIdx.x & 31;
  int row  = blockIdx.x * 8 + (threadIdx.x >> 5);
  if (row >= N) return;
  const float* xr = X + (long)row * 64;
  float d = xr[lane] * w[lane] + xr[32 + lane] * w[32 + lane];
#pragma unroll
  for (int off = 16; off > 0; off >>= 1) d += __shfl_xor(d, off, 32);
  if (lane == 0) out[row] = d + b[0];
}

// ---------------------------------------------------------------------------

static inline unsigned gceil(long n, int b) { return (unsigned)((n + b - 1) / b); }

struct SameP { const float *W, *as_, *ad_, *We, *ae, *b; };
struct BiP   { const float *Ws, *Wd, *as_, *ad_, *We, *ae, *b; };

extern "C" void kernel_launch(void* const* d_in, const int* in_sizes, int n_in,
                              void* d_out, int out_size, void* d_ws, size_t ws_size,
                              hipStream_t stream) {
  const float* x_p   = (const float*)d_in[0];
  const float* x_b   = (const float*)d_in[1];
  const float* ea_pp = (const float*)d_in[2];
  const float* ea_bp = (const float*)d_in[3];
  const float* ea_bb = (const float*)d_in[4];
  const int* src_pp  = (const int*)d_in[5];
  const int* dst_pp  = (const int*)d_in[6];
  const int* src_bp  = (const int*)d_in[7];
  const int* dst_bp  = (const int*)d_in[8];
  const int* src_bb  = (const int*)d_in[9];
  const int* dst_bb  = (const int*)d_in[10];

  const int NP  = in_sizes[0] / 32;
  const int NB  = in_sizes[1] / 16;
  const int EPP = in_sizes[5];
  const int EBP = in_sizes[7];
  const int EBB = in_sizes[9];

  // params flattened in setup_inputs() insertion order after the 11 data inputs
  const float** P = (const float**)(d_in + 11);
  const float *inpW = P[0], *inpB = P[1];
  const float *inbW = P[2], *inbB = P[3];
  const float *eppW = P[4], *eppB = P[5];
  const float *ebpW = P[6], *ebpB = P[7];
  const float *ebbW = P[8], *ebbB = P[9];
  // gat_same: W, as, ad, We, ae, b (6) ; gat_bi: Ws, Wd, as, ad, We, ae, b (7)
  SameP g_pp[2], g_bb[2]; BiP g_bp[2];
  int o = 10;
  for (int L = 0; L < 2; ++L) {
    g_pp[L] = SameP{P[o+0], P[o+1], P[o+2], P[o+3], P[o+4], P[o+5]}; o += 6;
    g_bp[L] = BiP  {P[o+0], P[o+1], P[o+2], P[o+3], P[o+4], P[o+5], P[o+6]}; o += 7;
    g_bb[L] = SameP{P[o+0], P[o+1], P[o+2], P[o+3], P[o+4], P[o+5]}; o += 6;
  }
  const float *outW = P[o], *outB = P[o + 1];

  // workspace layout (f32), ~109 MB total
  float* ws    = (float*)d_ws;
  float* xpA   = ws;
  float* xbA   = xpA  + (size_t)NP * 64;
  float* xpB   = xbA  + (size_t)NB * 64;
  float* xbB   = xpB  + (size_t)NP * 64;
  float* hbuf  = xbB  + (size_t)NB * 64;      // max(NP,NB) x 64
  float* asrc  = hbuf + (size_t)NB * 64;
  float* adst  = asrc + NB;
  float* mbuf  = adst + NB;
  float* sbuf  = mbuf + NB;
  float* ael   = sbuf + NB;
  float* logit = ael  + EBB;
  float* wae   = logit + EBB;
  float* wdad  = wae + 64;
  float* csum  = wdad + 64;

  // ---- input projections (leaky 0.01, bias) : WMMA bf16 ----
  k_gemm_n64<1, 1, true><<<gceil(NP, 128), 256, 0, stream>>>(x_p, inpW, inpB, xpA, NP, 32);
  k_gemm_n64<1, 1, true><<<gceil(NB, 128), 256, 0, stream>>>(x_b, inbW, inbB, xbA, NB, 16);

  auto conv_same = [&](const float* x, int n, const int* src, const int* dst, int E,
                       const float* EA, const float* Wp, const float* bp,
                       const SameP& g, float* xnext) {
    const float invE = 1.0f / (float)E;
    k_gemm_n64<2, 0, false><<<gceil(n, 128), 256, 0, stream>>>(x, g.W, nullptr, hbuf, n, 64);
    k_matvec64<<<1, 64, 0, stream>>>(g.We, g.ae, wae);
    k_rowdot2<<<gceil(n, 8), 256, 0, stream>>>(hbuf, g.as_, g.ad_, asrc, adst, n);
    k_zero<<<1, 256, 0, stream>>>(csum, 1);
    k_edge_ael<<<gceil(E, 256), 256, 0, stream>>>(EA, Wp, bp, wae, ael, csum, E, 12);
    k_node_init_same<<<gceil(n, 256), 256, 0, stream>>>(asrc, adst, csum, invE, mbuf, n);
    k_edge_logit_max<<<gceil(E, 256), 256, 0, stream>>>(src, dst, ael, asrc, adst, logit, mbuf, E);
    k_node_s_self<<<gceil(n, 256), 256, 0, stream>>>(asrc, adst, csum, invE, mbuf, sbuf, n);
    k_edge_expsum<<<gceil(E, 256), 256, 0, stream>>>(dst, logit, mbuf, sbuf, E);
    k_edge_coef<<<gceil(E, 256), 256, 0, stream>>>(dst, logit, mbuf, sbuf, E);
    k_edge_agg<<<gceil((long)E * 64, 256), 256, 0, stream>>>(src, dst, logit, hbuf, xnext, E);
    k_node_finish_same<<<gceil((long)n * 64, 256), 256, 0, stream>>>(
        hbuf, asrc, adst, csum, invE, mbuf, sbuf, g.b, xnext, n);
  };

  auto conv_bi = [&](const float* xs, int ns, const float* xd, int nd,
                     const int* src, const int* dst, int E,
                     const float* EA, const float* Wp, const float* bp,
                     const BiP& g, float* xnext) {
    k_gemm_n64<2, 0, false><<<gceil(ns, 128), 256, 0, stream>>>(xs, g.Ws, nullptr, hbuf, ns, 64);
    k_matvec64<<<1, 64, 0, stream>>>(g.We, g.ae, wae);
    k_matvec64<<<1, 64, 0, stream>>>(g.Wd, g.ad_, wdad);
    k_rowdot1<<<gceil(ns, 8), 256, 0, stream>>>(hbuf, g.as_, asrc, ns);
    k_rowdot1<<<gceil(nd, 8), 256, 0, stream>>>(xd, wdad, adst, nd);
    k_edge_ael<<<gceil(E, 256), 256, 0, stream>>>(EA, Wp, bp, wae, ael, nullptr, E, 12);
    k_node_init_bi<<<gceil(nd, 256), 256, 0, stream>>>(mbuf, sbuf, nd);
    k_edge_logit_max<<<gceil(E, 256), 256, 0, stream>>>(src, dst, ael, asrc, adst, logit, mbuf, E);
    k_edge_expsum<<<gceil(E, 256), 256, 0, stream>>>(dst, logit, mbuf, sbuf, E);
    k_edge_coef<<<gceil(E, 256), 256, 0, stream>>>(dst, logit, mbuf, sbuf, E);
    k_edge_agg<<<gceil((long)E * 64, 256), 256, 0, stream>>>(src, dst, logit, hbuf, xnext, E);
    k_node_bias<<<gceil((long)nd * 64, 256), 256, 0, stream>>>(g.b, xnext, nd);
  };

  float* xp  = xpA; float* xb  = xbA;
  float* xpn = xpB; float* xbn = xbB;
  for (int L = 0; L < 2; ++L) {
    k_zero<<<gceil((long)NP * 64, 256), 256, 0, stream>>>(xpn, (long)NP * 64);
    k_zero<<<gceil((long)NB * 64, 256), 256, 0, stream>>>(xbn, (long)NB * 64);
    conv_same(xp, NP, src_pp, dst_pp, EPP, ea_pp, eppW, eppB, g_pp[L], xpn);
    conv_bi  (xb, NB, xp, NP, src_bp, dst_bp, EBP, ea_bp, ebpW, ebpB, g_bp[L], xpn);
    conv_same(xb, NB, src_bb, dst_bb, EBB, ea_bb, ebbW, ebbB, g_bb[L], xbn);
    float* t;
    t = xp; xp = xpn; xpn = t;
    t = xb; xb = xbn; xbn = t;
  }

  k_out<<<gceil(NP, 8), 256, 0, stream>>>(xp, outW, outB, (float*)d_out, NP);
}
